// MultisourcesAnchoredCrossAttention_32581621908169
// MI455X (gfx1250) — compile-verified, matrix operand order checked by reading, not run on hardware
//
#include <hip/hip_runtime.h>

// ---------------------------------------------------------------------------
// MultisourcesAnchoredCrossAttention for MI455X (gfx1250, wave32, WMMA,
// async global->LDS staging)
// ---------------------------------------------------------------------------

typedef _Float16 v16h __attribute__((ext_vector_type(16)));
typedef float    v8f  __attribute__((ext_vector_type(8)));
typedef int      vec4i __attribute__((vector_size(16)));   // b128 payload type

#define B_SZ   4
#define HH_    128
#define WW_    128
#define DV_    768
#define DC_    128
#define INNER_ 1536
#define NH_    8
#define DH_    192
#define NTOK_  2048            // tokens per batch (2 * 1024 anchors)
#define MROWS_ 8192            // B * NTOK
#define KD_    384             // combined per-head feature (2*DH)

#if defined(__has_builtin)
#if __has_builtin(__builtin_amdgcn_global_load_async_to_lds_b128)
#define ASYNC_LDS_OK 1
#endif
#endif

// --------------------------- device helpers --------------------------------

__device__ __forceinline__ v8f zero8() {
  v8f z;
#pragma unroll
  for (int i = 0; i < 8; ++i) z[i] = 0.0f;
  return z;
}

// One 16-byte async copy global -> LDS (CDNA5 GLOBAL_LOAD_ASYNC_TO_LDS_B128,
// tracked by ASYNCcnt).  Fallback: plain load + ds_store.
__device__ __forceinline__ void cp_async_b128(_Float16* ldst,
                                              const _Float16* gsrc) {
#ifdef ASYNC_LDS_OK
  __builtin_amdgcn_global_load_async_to_lds_b128(
      (__attribute__((address_space(1))) vec4i*)const_cast<_Float16*>(gsrc),
      (__attribute__((address_space(3))) vec4i*)ldst, 0, 0);
#else
  *(uint4*)ldst = *(const uint4*)gsrc;
#endif
}

__device__ __forceinline__ void async_wait_all() {
#ifdef ASYNC_LDS_OK
#if __has_builtin(__builtin_amdgcn_s_wait_asynccnt)
  __builtin_amdgcn_s_wait_asynccnt(0);
#else
  asm volatile("s_wait_asynccnt 0" ::: "memory");
#endif
#endif
}

// Load a 16-lane A/B fragment for v_wmma_f32_16x16x32_f16.
// rowPtr points at the start of this lane's row (K-contiguous, f16).
// ISA layout (16-bit A 16x32): lanes 0-15 hold K = {kb+0..7, kb+16..23},
// lanes 16-31 hold K = {kb+8..15, kb+24..31}.  hi8 = (lane>>4)*8.
__device__ __forceinline__ v16h load_frag(const _Float16* __restrict__ rowPtr,
                                          int kb, int hi8) {
  const _Float16* p = rowPtr + kb + hi8;
  v16h f;
#pragma unroll
  for (int i = 0; i < 8; ++i) { f[i] = p[i]; f[8 + i] = p[16 + i]; }
  return f;
}

__device__ __forceinline__ v8f wmma32(v16h a, v16h b, v8f c) {
  // D = A(16x32) * B(32x16) + C, f32 accumulate
  return __builtin_amdgcn_wmma_f32_16x16x32_f16(
      false, a, false, b, (short)0, c, false, false);
}

// --------------------------- prep kernels ----------------------------------

__global__ __launch_bounds__(256) void gather_kernel(
    const float* __restrict__ values0, const float* __restrict__ values1,
    const float* __restrict__ coords0, const float* __restrict__ coords1,
    _Float16* __restrict__ av16, _Float16* __restrict__ ac16) {
  int row = blockIdx.x;             // b * NTOK + n
  int b = row >> 11;
  int n = row & (NTOK_ - 1);
  int s = n >> 10;                  // source 0/1
  int a = n & 1023;
  int r = (a >> 5) * 4 + 1;         // anchor row:  4*i + 1
  int c = (a & 31) * 4 + 1;         // anchor col:  4*j + 1
  const float* vs = (s == 0) ? values0 : values1;
  const float* cs = (s == 0) ? coords0 : coords1;
  const float* vsrc = vs + (((size_t)b * HH_ + r) * WW_ + c) * DV_;
  const float* csrc = cs + (((size_t)b * HH_ + r) * WW_ + c) * DC_;
  _Float16* vdst = av16 + (size_t)row * DV_;
  _Float16* cdst = ac16 + (size_t)row * DC_;
  for (int d = threadIdx.x; d < DV_; d += 256) vdst[d] = (_Float16)vsrc[d];
  for (int d = threadIdx.x; d < DC_; d += 256) cdst[d] = (_Float16)csrc[d];
}

// Transpose f32 (K x N) -> f16 (N x K)
__global__ __launch_bounds__(256) void transpose_convert_kernel(
    const float* __restrict__ src, _Float16* __restrict__ dst, int K, int N) {
  size_t idx = (size_t)blockIdx.x * 256 + threadIdx.x;
  size_t total = (size_t)K * N;
  if (idx >= total) return;
  int n = (int)(idx / K);
  int k = (int)(idx % K);
  dst[idx] = (_Float16)src[(size_t)k * N + n];
}

// --------------------------- WMMA GEMM (LDS double-buffered) ---------------
// C[M,N] = A[M,K] (f16 row-major) * Bt[N,K]^T (f16, N rows K-contig) [+ bias]
// Block tile 128x128, 256 threads = 8 waves (2x4), wave tile 64x32.
// A/B 128x32 tiles staged into LDS with async copies, double buffered.
// LDS row stride padded to 40 f16 (80B, 16B multiple) for bank spread.
template <typename OutT>
__global__ __launch_bounds__(256) void gemm_f16_kernel(
    const _Float16* __restrict__ A, const _Float16* __restrict__ Bt,
    OutT* __restrict__ C, const float* __restrict__ bias,
    int M, int N, int K) {
  __shared__ _Float16 ldsA[2][128 * 40];
  __shared__ _Float16 ldsB[2][128 * 40];

  int tid  = threadIdx.x;
  int w    = tid >> 5;
  int lane = tid & 31;
  int l16  = lane & 15;
  int hi   = lane >> 4;
  int hi8  = hi * 8;
  int mw   = w >> 2;                // 0..1
  int nw   = w & 3;                 // 0..3
  int bm   = blockIdx.y * 128;
  int bn   = blockIdx.x * 128;

  // stage one 128x32 A tile + one 128x32 B tile (1024 x 16B chunks)
  auto stage = [&](int kb, int buf) {
#pragma unroll
    for (int it = 0; it < 4; ++it) {
      int c = tid + it * 256;
      int isB = c >> 9;
      int cc = c & 511;
      int row = cc >> 2;            // 0..127
      int seg = cc & 3;             // 0..3 (8 f16 each)
      const _Float16* g =
          (isB ? Bt + (size_t)(bn + row) * K : A + (size_t)(bm + row) * K) +
          kb + seg * 8;
      _Float16* l = (isB ? ldsB[buf] : ldsA[buf]) + row * 40 + seg * 8;
      cp_async_b128(l, g);
    }
  };

  v8f acc[4][2];
#pragma unroll
  for (int i = 0; i < 4; ++i)
#pragma unroll
    for (int j = 0; j < 2; ++j) acc[i][j] = zero8();

  int nk = K / 32;
  stage(0, 0);
  for (int kt = 0; kt < nk; ++kt) {
    async_wait_all();
    __syncthreads();                      // tile kt ready in buf kt&1
    if (kt + 1 < nk) stage((kt + 1) * 32, (kt + 1) & 1);
    int buf = kt & 1;
    v16h af[4], bf[2];
#pragma unroll
    for (int i = 0; i < 4; ++i)
      af[i] = load_frag(ldsA[buf] + (mw * 64 + i * 16 + l16) * 40, 0, hi8);
#pragma unroll
    for (int j = 0; j < 2; ++j)
      bf[j] = load_frag(ldsB[buf] + (nw * 32 + j * 16 + l16) * 40, 0, hi8);
#pragma unroll
    for (int i = 0; i < 4; ++i)
#pragma unroll
      for (int j = 0; j < 2; ++j) acc[i][j] = wmma32(af[i], bf[j], acc[i][j]);
  }

  // C/D layout: col = lane&15, row = vgpr_r + 8*(lane>>4)
#pragma unroll
  for (int i = 0; i < 4; ++i) {
#pragma unroll
    for (int j = 0; j < 2; ++j) {
      int col = bn + nw * 32 + j * 16 + l16;
      float bv = (bias != nullptr) ? bias[col] : 0.0f;
#pragma unroll
      for (int r = 0; r < 8; ++r) {
        int rowi = bm + mw * 64 + i * 16 + r + 8 * hi;
        C[(size_t)rowi * N + col] = (OutT)(acc[i][j][r] + bv);
      }
    }
  }
}

// --------------------------- norm / pack kernel ----------------------------
// Reads qkv f16 (MROWS x 4608) and cpre f16 (MROWS x 3072).
// RMS-norms q,k over 1536 and coords over 3072; packs per-head:
//   Qcat[bh][n][0:192]   = qn * scale           (scale = 1/sqrt(192))
//   Qcat[bh][n][192:384] = qcn * relw * scale
//   Kcat[bh][n][0:192]   = kn
//   Kcat[bh][n][192:384] = kcn
//   Vt[bh][d][n]         = v
__global__ __launch_bounds__(256) void norm_pack_kernel(
    const _Float16* __restrict__ qkv, const _Float16* __restrict__ cpre,
    const float* __restrict__ qnorm_w, const float* __restrict__ knorm_w,
    const float* __restrict__ cnorm_w, const float* __restrict__ relw_p,
    _Float16* __restrict__ Qcat, _Float16* __restrict__ Kcat,
    _Float16* __restrict__ Vt) {
  __shared__ float sred[256];
  int row = blockIdx.x;
  int tid = threadIdx.x;
  int b = row >> 11;
  int n = row & (NTOK_ - 1);

  const _Float16* q  = qkv + (size_t)row * 4608;
  const _Float16* k  = q + 1536;
  const _Float16* v  = q + 3072;
  const _Float16* cp = cpre + (size_t)row * 3072;

  auto sumsq = [&](const _Float16* p, int len) -> float {
    float s = 0.0f;
    for (int i = tid; i < len; i += 256) { float x = (float)p[i]; s += x * x; }
    sred[tid] = s;
    __syncthreads();
    for (int off = 128; off > 0; off >>= 1) {
      if (tid < off) sred[tid] += sred[tid + off];
      __syncthreads();
    }
    float r = sred[0];
    __syncthreads();
    return r;
  };

  float rq = rsqrtf(sumsq(q, 1536) * (1.0f / 1536.0f) + 1e-6f);
  float rk = rsqrtf(sumsq(k, 1536) * (1.0f / 1536.0f) + 1e-6f);
  float rc = rsqrtf(sumsq(cp, 3072) * (1.0f / 3072.0f) + 1e-6f);

  float scale = rsqrtf((float)DH_);
  float relws = relw_p[0] * scale;       // COORDS_WEIGHT == 1.0

  for (int idx = tid; idx < 1536; idx += 256) {
    int h = idx / DH_;
    int d = idx % DH_;
    size_t bh = (size_t)(b * NH_ + h);
    size_t qo = (bh * NTOK_ + n) * KD_ + d;
    Qcat[qo]       = (_Float16)((float)q[idx] * rq * qnorm_w[idx] * scale);
    Qcat[qo + DH_] = (_Float16)((float)cp[idx] * rc * cnorm_w[idx] * relws);
    Kcat[qo]       = (_Float16)((float)k[idx] * rk * knorm_w[idx]);
    Kcat[qo + DH_] = (_Float16)((float)cp[1536 + idx] * rc * cnorm_w[1536 + idx]);
    Vt[(bh * DH_ + d) * NTOK_ + n] = v[idx];
  }
}

// --------------------------- flash attention -------------------------------
// grid = (B*NH = 32, NTOK/128 = 16), 256 threads = 8 waves.
// Each wave owns 16 queries; key loop in blocks of 32, K/V tiles staged in
// LDS by the whole block (async), P transposed through per-wave LDS.
__global__ __launch_bounds__(256) void flash_attn_kernel(
    const _Float16* __restrict__ Qcat, const _Float16* __restrict__ Kcat,
    const _Float16* __restrict__ Vt, _Float16* __restrict__ aout16) {
  __shared__ _Float16 ldsK[32 * 392];   // 32 keys x 384 (pad to 392)
  __shared__ _Float16 ldsV[192 * 40];   // 192 d    x 32 keys (pad to 40)
  __shared__ _Float16 Pl[8][16 * 32];   // per-wave P transpose scratch

  int bh   = blockIdx.x;
  int qblk = blockIdx.y;
  int tid  = threadIdx.x;
  int w    = tid >> 5;
  int lane = tid & 31;
  int l16  = lane & 15;
  int hi   = lane >> 4;
  int hi8  = hi * 8;
  int b    = bh >> 3;
  int h    = bh & 7;
  int q0   = qblk * 128 + w * 16;

  const _Float16* Qb = Qcat + (size_t)bh * NTOK_ * KD_;
  const _Float16* Kb = Kcat + (size_t)bh * NTOK_ * KD_;
  const _Float16* Vb = Vt   + (size_t)bh * DH_ * NTOK_;

  // Q fragments held in registers for all 12 k-steps (K=384)
  const _Float16* qrow = Qb + (size_t)(q0 + l16) * KD_;
  v16h qf[12];
#pragma unroll
  for (int c = 0; c < 12; ++c) qf[c] = load_frag(qrow, c * 32, hi8);

  v8f acc[12];
#pragma unroll
  for (int t = 0; t < 12; ++t) acc[t] = zero8();

  float mrun[8], lrun[8];
#pragma unroll
  for (int r = 0; r < 8; ++r) { mrun[r] = -3.0e38f; lrun[r] = 0.0f; }

  for (int kb = 0; kb < NTOK_; kb += 32) {
    // ---- stage K (32x384) + V (192x32) tiles: 2304 x 16B chunks ----
#pragma unroll
    for (int it = 0; it < 9; ++it) {
      int c = tid + it * 256;
      if (c < 1536) {                     // K tile
        int row = c / 48, seg = c % 48;
        cp_async_b128(ldsK + row * 392 + seg * 8,
                      Kb + (size_t)(kb + row) * KD_ + seg * 8);
      } else {                            // V tile
        int c2 = c - 1536;
        int row = c2 >> 2, seg = c2 & 3;
        cp_async_b128(ldsV + row * 40 + seg * 8,
                      Vb + (size_t)row * NTOK_ + kb + seg * 8);
      }
    }
    async_wait_all();
    __syncthreads();

    // ---- scores: 2 key subtiles of 16, K-dim 384 ----
    v8f s[2];
#pragma unroll
    for (int sub = 0; sub < 2; ++sub) s[sub] = zero8();
#pragma unroll
    for (int sub = 0; sub < 2; ++sub) {
      const _Float16* krow = ldsK + (sub * 16 + l16) * 392;
#pragma unroll
      for (int c = 0; c < 12; ++c) {
        v16h kf = load_frag(krow, c * 32, hi8);
        s[sub] = wmma32(qf[c], kf, s[sub]);
      }
    }

    // ---- online softmax update ----
    float alpha[8];
#pragma unroll
    for (int r = 0; r < 8; ++r) {
      float mb = fmaxf(s[0][r], s[1][r]);
#pragma unroll
      for (int msk = 1; msk < 16; msk <<= 1)
        mb = fmaxf(mb, __shfl_xor(mb, msk));        // reduce 16-lane half
      float mnew = fmaxf(mrun[r], mb);
      alpha[r] = __expf(mrun[r] - mnew);
      float ps = 0.0f;
#pragma unroll
      for (int sub = 0; sub < 2; ++sub) {
        float pv = __expf(s[sub][r] - mnew);
        ps += pv;
        Pl[w][(r + 8 * hi) * 32 + sub * 16 + l16] = (_Float16)pv;
      }
#pragma unroll
      for (int msk = 1; msk < 16; msk <<= 1) ps += __shfl_xor(ps, msk);
      lrun[r] = lrun[r] * alpha[r] + ps;
      mrun[r] = mnew;
    }

    // rescale running output
#pragma unroll
    for (int t = 0; t < 12; ++t)
#pragma unroll
      for (int r = 0; r < 8; ++r) acc[t][r] *= alpha[r];

    // ---- P (16x32, via LDS transpose) x V (32x192) ----
    v16h pf = load_frag(&Pl[w][l16 * 32], 0, hi8);
#pragma unroll
    for (int t = 0; t < 12; ++t) {
      v16h vf = load_frag(ldsV + (t * 16 + l16) * 40, 0, hi8);
      acc[t] = wmma32(pf, vf, acc[t]);
    }
    __syncthreads();                      // done reading before next stage
  }

  // ---- epilogue: normalize, write f16 attention output ----
  float inv[8];
#pragma unroll
  for (int r = 0; r < 8; ++r) inv[r] = 1.0f / lrun[r];
#pragma unroll
  for (int t = 0; t < 12; ++t) {
    int feat = h * DH_ + t * 16 + l16;
#pragma unroll
    for (int r = 0; r < 8; ++r) {
      int nq = q0 + r + 8 * hi;
      aout16[((size_t)(b * NTOK_ + nq)) * INNER_ + feat] =
          (_Float16)(acc[t][r] * inv[r]);
    }
  }
}

// --------------------------- scatter-add -----------------------------------
__global__ __launch_bounds__(256) void scatter_add_kernel(
    const float* __restrict__ proj, float* __restrict__ out) {
  int row = blockIdx.x;             // b * NTOK + n
  int b = row >> 11;
  int n = row & (NTOK_ - 1);
  int s = n >> 10;
  int a = n & 1023;
  int r = (a >> 5) * 4 + 1;
  int c = (a & 31) * 4 + 1;
  float* dst = out + ((((size_t)s * B_SZ + b) * HH_ + r) * WW_ + c) * DV_;
  const float* src = proj + (size_t)row * DV_;
  for (int d = threadIdx.x; d < DV_; d += 256) dst[d] += src[d];
}

// --------------------------- host launcher ---------------------------------

extern "C" void kernel_launch(void* const* d_in, const int* in_sizes, int n_in,
                              void* d_out, int out_size, void* d_ws,
                              size_t ws_size, hipStream_t stream) {
  (void)in_sizes; (void)n_in; (void)out_size; (void)ws_size;

  const float* values0  = (const float*)d_in[0];
  const float* coords0  = (const float*)d_in[1];
  const float* values1  = (const float*)d_in[2];
  const float* coords1  = (const float*)d_in[3];
  const float* w_qkv    = (const float*)d_in[4];
  const float* qnorm_w  = (const float*)d_in[5];
  const float* knorm_w  = (const float*)d_in[6];
  const float* w_coords = (const float*)d_in[7];
  const float* cnorm_w  = (const float*)d_in[8];
  const float* relw     = (const float*)d_in[9];
  const float* out_w    = (const float*)d_in[10];
  const float* out_b    = (const float*)d_in[11];
  float* out = (float*)d_out;

  char* ws = (char*)d_ws;
  size_t off = 0;
  auto alloc = [&](size_t bytes) -> size_t {
    size_t r = off;
    off += (bytes + 255) & ~(size_t)255;
    return r;
  };

  size_t o_av16 = alloc((size_t)MROWS_ * DV_ * 2);       // gathered values f16
  size_t o_ac16 = alloc((size_t)MROWS_ * DC_ * 2);       // gathered coords f16
  size_t o_wq16 = alloc((size_t)4608 * 768 * 2);         // w_qkv^T
  size_t o_wc16 = alloc((size_t)3072 * 128 * 2);         // w_coords^T
  size_t o_wo16 = alloc((size_t)768 * 1536 * 2);         // out_w^T
  size_t o_qkv  = alloc((size_t)MROWS_ * 4608 * 2);      // qkv f16
  size_t o_cpre = alloc((size_t)MROWS_ * 3072 * 2);      // coords proj f16
  size_t o_Qcat = alloc((size_t)B_SZ * NH_ * NTOK_ * KD_ * 2);
  size_t o_Kcat = alloc((size_t)B_SZ * NH_ * NTOK_ * KD_ * 2);
  size_t o_Vt   = alloc((size_t)B_SZ * NH_ * DH_ * NTOK_ * 2);
  // qkv region is dead after norm_pack -> alias attn output + projection
  size_t o_aout = o_qkv;                                 // f16, MROWS x 1536
  size_t o_proj = o_qkv + (((size_t)MROWS_ * INNER_ * 2 + 255) & ~(size_t)255);

  _Float16* av16 = (_Float16*)(ws + o_av16);
  _Float16* ac16 = (_Float16*)(ws + o_ac16);
  _Float16* wq16 = (_Float16*)(ws + o_wq16);
  _Float16* wc16 = (_Float16*)(ws + o_wc16);
  _Float16* wo16 = (_Float16*)(ws + o_wo16);
  _Float16* qkv  = (_Float16*)(ws + o_qkv);
  _Float16* cpre = (_Float16*)(ws + o_cpre);
  _Float16* Qcat = (_Float16*)(ws + o_Qcat);
  _Float16* Kcat = (_Float16*)(ws + o_Kcat);
  _Float16* Vt   = (_Float16*)(ws + o_Vt);
  _Float16* aout = (_Float16*)(ws + o_aout);
  float*    proj = (float*)(ws + o_proj);

  // 1. residual base: out = stack(values0, values1)
  size_t valbytes = (size_t)B_SZ * HH_ * WW_ * DV_ * sizeof(float);
  (void)hipMemcpyAsync(out, values0, valbytes, hipMemcpyDeviceToDevice, stream);
  (void)hipMemcpyAsync((char*)out + valbytes, values1, valbytes,
                       hipMemcpyDeviceToDevice, stream);

  // 2. gather anchors -> f16
  gather_kernel<<<MROWS_, 256, 0, stream>>>(values0, values1, coords0, coords1,
                                            av16, ac16);

  // 3. weight transposes -> f16 [N][K]
  {
    size_t t1 = (size_t)768 * 4608;
    transpose_convert_kernel<<<(unsigned)((t1 + 255) / 256), 256, 0, stream>>>(
        w_qkv, wq16, 768, 4608);
    size_t t2 = (size_t)128 * 3072;
    transpose_convert_kernel<<<(unsigned)((t2 + 255) / 256), 256, 0, stream>>>(
        w_coords, wc16, 128, 3072);
    size_t t3 = (size_t)1536 * 768;
    transpose_convert_kernel<<<(unsigned)((t3 + 255) / 256), 256, 0, stream>>>(
        out_w, wo16, 1536, 768);
  }

  // 4. qkv = av @ w_qkv          (8192 x 4608, K=768) -> f16
  gemm_f16_kernel<_Float16><<<dim3(4608 / 128, MROWS_ / 128), 256, 0, stream>>>(
      av16, wq16, qkv, nullptr, MROWS_, 4608, 768);

  // 5. cpre = ac @ w_coords      (8192 x 3072, K=128) -> f16
  gemm_f16_kernel<_Float16><<<dim3(3072 / 128, MROWS_ / 128), 256, 0, stream>>>(
      ac16, wc16, cpre, nullptr, MROWS_, 3072, 128);

  // 6. RMS norms + per-head packing (folds softmax scale & rel weight)
  norm_pack_kernel<<<MROWS_, 256, 0, stream>>>(qkv, cpre, qnorm_w, knorm_w,
                                               cnorm_w, relw, Qcat, Kcat, Vt);

  // 7. flash attention -> aout f16 (8192 x 1536)
  flash_attn_kernel<<<dim3(B_SZ * NH_, NTOK_ / 128), 256, 0, stream>>>(
      Qcat, Kcat, Vt, aout);

  // 8. proj = aout @ out_w + out_b   (8192 x 768, K=1536) -> f32
  gemm_f16_kernel<float><<<dim3(768 / 128, MROWS_ / 128), 256, 0, stream>>>(
      aout, wo16, proj, out_b, MROWS_, 768, 1536);

  // 9. scatter-add into residual copies
  scatter_add_kernel<<<MROWS_, 256, 0, stream>>>(proj, out);
}
